// MultiHeadAttentionAd_85031762526627
// MI455X (gfx1250) — matrix operand order repaired and checked by reading, not compiled, not run on hardware
//
#include <hip/hip_runtime.h>

typedef __attribute__((ext_vector_type(16))) _Float16 v16h;
typedef __attribute__((ext_vector_type(8)))  float    v8f;
typedef __attribute__((ext_vector_type(4)))  unsigned int u32x4;
typedef __attribute__((ext_vector_type(8)))  int      i32x8;
typedef __attribute__((ext_vector_type(4)))  int      i32x4;

#if defined(__AMDGCN__) && __has_builtin(__builtin_amdgcn_tensor_load_to_lds) && \
    __has_builtin(__builtin_amdgcn_s_wait_tensorcnt)
#define USE_TDM 1
#else
#define USE_TDM 0
#endif

namespace {

constexpr int kB    = 8;
constexpr int kS    = 2048;
constexpr int kDin  = 1024;
constexpr int kDout = 1024;
constexpr int TM = 128, TN = 128, TK = 32;
constexpr int LDT = TK + 2;           // padded LDS row stride (f16 elems) = 17 DWORDs
constexpr float kScale = 0.03125f;    // 1/sqrt(1024)

union Frag { v16h v; unsigned u[8]; };
union H8   { uint4 q; _Float16 h[8]; };

__device__ inline unsigned pack2h(float a, float b) {
  union { _Float16 h[2]; unsigned u; } p;
  p.h[0] = (_Float16)a; p.h[1] = (_Float16)b;
  return p.u;
}

// A fragment: 16x32 f16, row-major tile in LDS (row stride LDT).
// Lane L: M = L&15, upper half-wave shifts K by 8. VGPR i holds K pair
// base = (i<4 ? 2i : 16+2(i-4)) + 8*(L>>4)   (per ISA 7.12.2).
__device__ inline v16h gather_A(const _Float16* s) {
  const int lane = threadIdx.x & 31;
  const int m  = lane & 15;
  const int h8 = (lane >> 4) << 3;
  Frag f;
#pragma unroll
  for (int i = 0; i < 8; ++i) {
    const int kb = ((i < 4) ? (2 * i) : (16 + 2 * (i - 4))) + h8;
    f.u[i] = *(const unsigned*)(s + m * LDT + kb);
  }
  return f.v;
}

// B fragment: 32x16 f16. LDS tile stored TRANSPOSED: sB[n][k], so each lane's
// K-pairs are contiguous. Lane L: N = L&15, K base = (i<4 ? 2i : 16+2(i-4)) + 8*(L>>4).
__device__ inline v16h gather_B(const _Float16* s) {
  const int lane = threadIdx.x & 31;
  const int n   = lane & 15;
  const int kb0 = (lane >> 4) << 3;
  Frag f;
#pragma unroll
  for (int i = 0; i < 8; ++i)
    f.u[i] = *(const unsigned*)(s + n * LDT + kb0 + ((i < 4) ? (2 * i) : (16 + 2 * (i - 4))));
  return f.v;
}

__device__ inline v8f wmma_f16(v16h a, v16h b, v8f c) {
  return __builtin_amdgcn_wmma_f32_16x16x32_f16(false, a, false, b,
                                                (short)0, c, false, false);
}

#if USE_TDM
// TDM: DMA a 128-row x 32-col f16 tile (row stride = kDout elems) from global
// into LDS, inserting 1 pad DWORD after every 16 DWORDs (64B row) so the LDS
// pitch is 17 DWORDs = LDT(34) f16 elems. Descriptor per ISA cdna5 ch8.
__device__ inline void tdm_tile_load(unsigned lds_byte, const _Float16* g) {
  const unsigned long long ga = (unsigned long long)(size_t)g;
  const u32x4 g0 = {
      1u,                                          // count=1, no gather
      lds_byte,                                    // lds_addr
      (unsigned)(ga & 0xFFFFFFFFu),                // global_addr[31:0]
      (unsigned)((ga >> 32) & 0x01FFFFFFu) | 0x80000000u  // addr[56:32] | type=2
  };
  const i32x8 g1 = {
      (int)((1u << 16) | (1u << 20) | (3u << 22)), // data_size=2B, pad_en, interval=16DW (+1DW)
      (int)((unsigned)kDout << 16),                // tensor_dim0 = 1024 (bits 79:48 lo)
      (int)((unsigned)kS << 16),                   // tensor_dim1 = 2048 (bits 111:80 lo)
      (int)(32u << 16),                            // tile_dim0 = 32 (bits 127:112)
      128,                                         // tile_dim1 = 128
      kDout,                                       // tensor_dim0_stride = 1024 elems
      0, 0
  };
  const i32x4 z4 = (i32x4)0;                       // 2-D tile: groups 2/3 unused
  const i32x8 z8 = (i32x8)0;                       // extra group (6-arg toolchain form)
  __builtin_amdgcn_tensor_load_to_lds(g0, g1, z4, z4, z8, 0);
}
#endif

} // namespace

// ---------------------------------------------------------------------------
// Kernel 1: P[M,1024] = X[M,1024](f32) * W[1024,1024](f32) + bias, output f16.
// grid = (N/128, M/128), block = 256 (8 waves, 2x4 wave grid, 64x32 per wave).
// ---------------------------------------------------------------------------
__global__ __launch_bounds__(256) void k_proj_gemm(
    const float* __restrict__ X, const float* __restrict__ W,
    const float* __restrict__ bias, _Float16* __restrict__ P) {
  __shared__ __align__(16) _Float16 sA[TM * LDT];
  __shared__ __align__(16) _Float16 sB[TN * LDT];   // transposed [n][k]
  const int tid  = threadIdx.x;
  const int lane = tid & 31;
  const int wave = tid >> 5;
  const int wm = wave >> 2, wn = wave & 3;
  const int m0 = blockIdx.y * TM, n0 = blockIdx.x * TN;

  v8f acc[4][2];
#pragma unroll
  for (int mt = 0; mt < 4; ++mt)
#pragma unroll
    for (int nt = 0; nt < 2; ++nt) acc[mt][nt] = (v8f)0.0f;

  for (int k0 = 0; k0 < kDin; k0 += TK) {
    { // stage A: 128x32 f32 -> f16, row-major
      const int r = tid >> 3;
      const int c = (tid & 7) * 4;
#pragma unroll
      for (int j = 0; j < 4; ++j) {
        const int row = r + 32 * j;
        const float4 f = *(const float4*)(X + (size_t)(m0 + row) * kDin + k0 + c);
        unsigned* d = (unsigned*)(sA + row * LDT + c);
        d[0] = pack2h(f.x, f.y);
        d[1] = pack2h(f.z, f.w);
      }
      if (k0 + TK < kDin)   // prefetch next A line -> global_prefetch_b8
        __builtin_prefetch(X + (size_t)(m0 + r) * kDin + k0 + TK + c, 0, 1);
    }
    { // stage B transposed: sB[n][k] = f16(W[k0+k][n0+n])
      const int k  = tid & 31;
      const int n4 = tid >> 5;
#pragma unroll
      for (int j = 0; j < 4; ++j) {
        const int n = (n4 + 8 * j) * 4;
        const float4 f = *(const float4*)(W + (size_t)(k0 + k) * kDout + n0 + n);
        sB[(n + 0) * LDT + k] = (_Float16)f.x;
        sB[(n + 1) * LDT + k] = (_Float16)f.y;
        sB[(n + 2) * LDT + k] = (_Float16)f.z;
        sB[(n + 3) * LDT + k] = (_Float16)f.w;
      }
      if (k0 + TK < kDin)   // prefetch next B line
        __builtin_prefetch(W + (size_t)(k0 + TK + k) * kDout + n0 + n4 * 4, 0, 1);
    }
    __syncthreads();
    const _Float16* Ab = sA + (wm * 64) * LDT;
    const _Float16* Bb = sB + (wn * 32) * LDT;
    const v16h b0 = gather_B(Bb);
    const v16h b1 = gather_B(Bb + 16 * LDT);
#pragma unroll
    for (int mt = 0; mt < 4; ++mt) {
      const v16h a = gather_A(Ab + mt * 16 * LDT);
      acc[mt][0] = wmma_f16(a, b0, acc[mt][0]);
      acc[mt][1] = wmma_f16(a, b1, acc[mt][1]);
    }
    __syncthreads();
  }

  const int cn = lane & 15;
  const int rh = (lane >> 4) * 8;
#pragma unroll
  for (int mt = 0; mt < 4; ++mt)
#pragma unroll
    for (int nt = 0; nt < 2; ++nt) {
      const int col = n0 + wn * 32 + nt * 16 + cn;
      const float bv = bias[col];
#pragma unroll
      for (int r = 0; r < 8; ++r) {
        const int row = m0 + wm * 64 + mt * 16 + rh + r;
        P[(size_t)row * kDout + col] = (_Float16)(acc[mt][nt][r] + bv);
      }
    }
}

// ---------------------------------------------------------------------------
// Kernel 2: scores[b] = Q[b] * K[b]^T * scale, mask -> -1e9, f32 out.
// A/B tiles are straight f16 copies -> staged by the Tensor Data Mover with
// double buffering: wave 0 issues tensor_load_to_lds for step ks+1 while all
// waves run WMMA on step ks, then s_wait_tensorcnt + workgroup barrier.
// grid = (S/128, S/128, B)
// ---------------------------------------------------------------------------
__global__ __launch_bounds__(256) void k_scores_gemm(
    const _Float16* __restrict__ Q, const _Float16* __restrict__ K,
    const int* __restrict__ mask, float* __restrict__ attn) {
  __shared__ __align__(16) _Float16 sA[2][TM * LDT];
  __shared__ __align__(16) _Float16 sB[2][TN * LDT];
  const int tid  = threadIdx.x;
  const int lane = tid & 31;
  const int wave = tid >> 5;
  const int wm = wave >> 2, wn = wave & 3;
  const int m0 = blockIdx.y * TM, n0 = blockIdx.x * TN;
  const int b  = blockIdx.z;
  const _Float16* Qb = Q + (size_t)b * kS * kDout;
  const _Float16* Kb = K + (size_t)b * kS * kDout;
  const int*      Mb = mask + (size_t)b * kS * kS;
  float*          Sb = attn + (size_t)b * kS * kS;

  v8f acc[4][2];
#pragma unroll
  for (int mt = 0; mt < 4; ++mt)
#pragma unroll
    for (int nt = 0; nt < 2; ++nt) acc[mt][nt] = (v8f)0.0f;

  constexpr int NK = kDout / TK;   // 32 k-steps

#if USE_TDM
  const unsigned ldsA[2] = { (unsigned)(size_t)&sA[0][0],
                             (unsigned)(size_t)&sA[1][0] };
  const unsigned ldsB[2] = { (unsigned)(size_t)&sB[0][0],
                             (unsigned)(size_t)&sB[1][0] };
  if (wave == 0) {      // preload k-step 0 into buffer 0
    tdm_tile_load(ldsA[0], Qb + (size_t)m0 * kDout);
    tdm_tile_load(ldsB[0], Kb + (size_t)n0 * kDout);
    __builtin_amdgcn_s_wait_tensorcnt(0);
  }
  __syncthreads();
  for (int ks = 0; ks < NK; ++ks) {
    const int cur = ks & 1;
    if (wave == 0 && ks + 1 < NK) {       // async-stage next step into alt buffer
      const int kn = (ks + 1) * TK;
      tdm_tile_load(ldsA[cur ^ 1], Qb + (size_t)m0 * kDout + kn);
      tdm_tile_load(ldsB[cur ^ 1], Kb + (size_t)n0 * kDout + kn);
    }
    const _Float16* Ab = &sA[cur][0] + (wm * 64) * LDT;
    const _Float16* Bb = &sB[cur][0] + (wn * 32) * LDT;
    const v16h b0 = gather_B(Bb);
    const v16h b1 = gather_B(Bb + 16 * LDT);
#pragma unroll
    for (int mt = 0; mt < 4; ++mt) {
      const v16h a = gather_A(Ab + mt * 16 * LDT);
      acc[mt][0] = wmma_f16(a, b0, acc[mt][0]);
      acc[mt][1] = wmma_f16(a, b1, acc[mt][1]);
    }
    if (wave == 0) __builtin_amdgcn_s_wait_tensorcnt(0);
    __syncthreads();
  }
#else
  for (int ks = 0; ks < NK; ++ks) {
    const int k0 = ks * TK;
    { // manual stage: A tile = Q rows, B tile = K rows (f16 copies)
      const int r  = tid >> 1;
      const int gb = (tid & 1) * 2;
#pragma unroll
      for (int j = 0; j < 2; ++j) {
        const int c8 = (gb + j) * 8;
        const uint4 ua = *(const uint4*)(Qb + (size_t)(m0 + r) * kDout + k0 + c8);
        unsigned* da = (unsigned*)(&sA[0][0] + r * LDT + c8);
        da[0] = ua.x; da[1] = ua.y; da[2] = ua.z; da[3] = ua.w;
        const uint4 ub = *(const uint4*)(Kb + (size_t)(n0 + r) * kDout + k0 + c8);
        unsigned* db = (unsigned*)(&sB[0][0] + r * LDT + c8);
        db[0] = ub.x; db[1] = ub.y; db[2] = ub.z; db[3] = ub.w;
      }
    }
    __syncthreads();
    const _Float16* Ab = &sA[0][0] + (wm * 64) * LDT;
    const _Float16* Bb = &sB[0][0] + (wn * 32) * LDT;
    const v16h b0 = gather_B(Bb);
    const v16h b1 = gather_B(Bb + 16 * LDT);
#pragma unroll
    for (int mt = 0; mt < 4; ++mt) {
      const v16h a = gather_A(Ab + mt * 16 * LDT);
      acc[mt][0] = wmma_f16(a, b0, acc[mt][0]);
      acc[mt][1] = wmma_f16(a, b1, acc[mt][1]);
    }
    __syncthreads();
  }
#endif

  const int cn = lane & 15;
  const int rh = (lane >> 4) * 8;
#pragma unroll
  for (int mt = 0; mt < 4; ++mt)
#pragma unroll
    for (int nt = 0; nt < 2; ++nt) {
      const int col = n0 + wn * 32 + nt * 16 + cn;
#pragma unroll
      for (int r = 0; r < 8; ++r) {
        const int row = m0 + wm * 64 + mt * 16 + rh + r;
        const size_t idx = (size_t)row * kS + col;
        float s = acc[mt][nt][r] * kScale;
        if (Mb[idx] != 0) s = -1e9f;
        Sb[idx] = s;
      }
    }
}

// ---------------------------------------------------------------------------
// Kernel 3: in-place row softmax over 2048-wide rows. grid = B*S, block = 256.
// ---------------------------------------------------------------------------
__global__ __launch_bounds__(256) void k_softmax(float* __restrict__ attn) {
  float* p = attn + (size_t)blockIdx.x * kS;
  const int t = threadIdx.x;
  float v[8];
  float mx = -3.4e38f;
#pragma unroll
  for (int i = 0; i < 8; ++i) { v[i] = p[t + 256 * i]; mx = fmaxf(mx, v[i]); }
#pragma unroll
  for (int off = 16; off > 0; off >>= 1) mx = fmaxf(mx, __shfl_xor(mx, off, 32));
  __shared__ float redm[8];
  __shared__ float reds[8];
  if ((t & 31) == 0) redm[t >> 5] = mx;
  __syncthreads();
  mx = redm[0];
#pragma unroll
  for (int i = 1; i < 8; ++i) mx = fmaxf(mx, redm[i]);
  float sum = 0.0f;
#pragma unroll
  for (int i = 0; i < 8; ++i) { v[i] = __expf(v[i] - mx); sum += v[i]; }
#pragma unroll
  for (int off = 16; off > 0; off >>= 1) sum += __shfl_xor(sum, off, 32);
  if ((t & 31) == 0) reds[t >> 5] = sum;
  __syncthreads();
  sum = 0.0f;
#pragma unroll
  for (int i = 0; i < 8; ++i) sum += reds[i];
  const float inv = 1.0f / sum;
#pragma unroll
  for (int i = 0; i < 8; ++i) p[t + 256 * i] = v[i] * inv;
}

// ---------------------------------------------------------------------------
// Kernel 4: out[b] = attn[b](f32, SxS) * V[b](f16, Sx1024), f32 out.
// grid = (1024/128, S/128, B)
// ---------------------------------------------------------------------------
__global__ __launch_bounds__(256) void k_av_gemm(
    const float* __restrict__ attn, const _Float16* __restrict__ V,
    float* __restrict__ out) {
  __shared__ __align__(16) _Float16 sA[TM * LDT];
  __shared__ __align__(16) _Float16 sB[TN * LDT];
  const int tid  = threadIdx.x;
  const int lane = tid & 31;
  const int wave = tid >> 5;
  const int wm = wave >> 2, wn = wave & 3;
  const int m0 = blockIdx.y * TM, n0 = blockIdx.x * TN;
  const int b  = blockIdx.z;
  const float*    Ab_g = attn + (size_t)b * kS * kS;
  const _Float16* Vb   = V    + (size_t)b * kS * kDout;
  float*          Ob   = out  + (size_t)b * kS * kDout;

  v8f acc[4][2];
#pragma unroll
  for (int mt = 0; mt < 4; ++mt)
#pragma unroll
    for (int nt = 0; nt < 2; ++nt) acc[mt][nt] = (v8f)0.0f;

  for (int k0 = 0; k0 < kS; k0 += TK) {
    { // stage A: attn weights f32 -> f16, lda = kS
      const int r = tid >> 3;
      const int c = (tid & 7) * 4;
#pragma unroll
      for (int j = 0; j < 4; ++j) {
        const int row = r + 32 * j;
        const float4 f = *(const float4*)(Ab_g + (size_t)(m0 + row) * kS + k0 + c);
        unsigned* d = (unsigned*)(sA + row * LDT + c);
        d[0] = pack2h(f.x, f.y);
        d[1] = pack2h(f.z, f.w);
      }
      if (k0 + TK < kS)
        __builtin_prefetch(Ab_g + (size_t)(m0 + r) * kS + k0 + TK + c, 0, 1);
    }
    { // stage B transposed: sB[n][k] = V[k0+k][n0+n]
      const int k  = tid & 31;
      const int n8 = tid >> 5;
#pragma unroll
      for (int j = 0; j < 2; ++j) {
        const int n = (n8 + 8 * j) * 8;
        H8 u; u.q = *(const uint4*)(Vb + (size_t)(k0 + k) * kDout + n0 + n);
#pragma unroll
        for (int i = 0; i < 8; ++i) sB[(n + i) * LDT + k] = u.h[i];
      }
      if (k0 + TK < kS)
        __builtin_prefetch(Vb + (size_t)(k0 + TK + k) * kDout + n0 + n8 * 8, 0, 1);
    }
    __syncthreads();
    const _Float16* Abs = sA + (wm * 64) * LDT;
    const _Float16* Bbs = sB + (wn * 32) * LDT;
    const v16h b0 = gather_B(Bbs);
    const v16h b1 = gather_B(Bbs + 16 * LDT);
#pragma unroll
    for (int mt = 0; mt < 4; ++mt) {
      const v16h a = gather_A(Abs + mt * 16 * LDT);
      acc[mt][0] = wmma_f16(a, b0, acc[mt][0]);
      acc[mt][1] = wmma_f16(a, b1, acc[mt][1]);
    }
    __syncthreads();
  }

  const int cn = lane & 15;
  const int rh = (lane >> 4) * 8;
#pragma unroll
  for (int mt = 0; mt < 4; ++mt)
#pragma unroll
    for (int nt = 0; nt < 2; ++nt) {
      const int col = n0 + wn * 32 + nt * 16 + cn;
#pragma unroll
      for (int r = 0; r < 8; ++r) {
        const int row = m0 + wm * 64 + mt * 16 + rh + r;
        Ob[(size_t)row * kDout + col] = acc[mt][nt][r];
      }
    }
}

// ---------------------------------------------------------------------------
extern "C" void kernel_launch(void* const* d_in, const int* in_sizes, int n_in,
                              void* d_out, int out_size, void* d_ws, size_t ws_size,
                              hipStream_t stream) {
  (void)in_sizes; (void)n_in; (void)out_size; (void)ws_size;
  const float* q    = (const float*)d_in[0];
  const float* k    = (const float*)d_in[1];
  const float* v    = (const float*)d_in[2];
  const int*   mask = (const int*)d_in[3];
  const float* Wq   = (const float*)d_in[4];
  const float* bq   = (const float*)d_in[5];
  const float* Wk   = (const float*)d_in[6];
  const float* bk   = (const float*)d_in[7];
  const float* Wv   = (const float*)d_in[8];
  const float* bv   = (const float*)d_in[9];

  float* out  = (float*)d_out;                       // [B,S,Dout]
  float* attn = out + (size_t)kB * kS * kDout;       // [B,S,S] (second output)

  const size_t per = (size_t)kB * kS * kDout;        // elements per f16 buffer
  _Float16* Qh = (_Float16*)d_ws;                    // 3 * 32 MB f16 in workspace
  _Float16* Kh = Qh + per;
  _Float16* Vh = Kh + per;

  const dim3 blk(256);
  const dim3 gproj(kDout / TN, (kB * kS) / TM);      // (8, 128)
  k_proj_gemm<<<gproj, blk, 0, stream>>>(q, Wq, bq, Qh);
  k_proj_gemm<<<gproj, blk, 0, stream>>>(k, Wk, bk, Kh);
  k_proj_gemm<<<gproj, blk, 0, stream>>>(v, Wv, bv, Vh);

  k_scores_gemm<<<dim3(kS / TN, kS / TM, kB), blk, 0, stream>>>(Qh, Kh, mask, attn);
  k_softmax<<<dim3(kB * kS), blk, 0, stream>>>(attn);
  k_av_gemm<<<dim3(kDout / TN, kS / TM, kB), blk, 0, stream>>>(attn, Vh, out);
}
